// Attention_14577119003164
// MI455X (gfx1250) — matrix operand order repaired and checked by reading, not compiled, not run on hardware
//
#include <hip/hip_runtime.h>

#define NMOD 6
#define BATCH 8192
#define DIM 2048

typedef __attribute__((ext_vector_type(16))) __bf16 v16bf;
typedef __attribute__((ext_vector_type(8)))  float  v8f;

union Frag {
    v16bf v;
    unsigned short u[16];
    unsigned int   d[8];
};

static __device__ __forceinline__ unsigned short f2bf(float f) {
    union { float f; unsigned int u; } a;
    a.f = f;
    unsigned int r = a.u + 0x7FFFu + ((a.u >> 16) & 1u); // round-to-nearest-even
    return (unsigned short)(r >> 16);
}

// ---------------------------------------------------------------------------
// Kernel 1: scaler[b] = 1 + #modalities whose row x[m,b,:] sums to exactly 0
// ---------------------------------------------------------------------------
__global__ __launch_bounds__(256) void scaler_kernel(const float* __restrict__ x,
                                                     float* __restrict__ scaler) {
    __shared__ float red[256];
    const int b = blockIdx.x;
    const int tid = threadIdx.x;
    int zeros = 0;
    for (int m = 0; m < NMOD; ++m) {
        const float* row = x + ((size_t)(m * BATCH + b)) * DIM;
        float s = 0.f;
        for (int i = tid; i < DIM; i += 256) s += row[i];
        red[tid] = s;
        __syncthreads();
        for (int off = 128; off > 0; off >>= 1) {
            if (tid < off) red[tid] += red[tid + off];
            __syncthreads();
        }
        if (tid == 0 && red[0] == 0.0f) zeros++;
        __syncthreads();
    }
    if (tid == 0) scaler[b] = (float)(zeros + 1);
}

// ---------------------------------------------------------------------------
// Kernel 2: Wb[i] = bf16(W[i])   (straight convert, keeps [e][d] row-major)
// ---------------------------------------------------------------------------
__global__ __launch_bounds__(256) void wconv_kernel(const float* __restrict__ W,
                                                    unsigned short* __restrict__ Wb) {
    const int i = (blockIdx.x * 256 + threadIdx.x) * 4;
    float4 v4 = *(const float4*)(W + i);
    unsigned int u01 = (unsigned int)f2bf(v4.x) | ((unsigned int)f2bf(v4.y) << 16);
    unsigned int u23 = (unsigned int)f2bf(v4.z) | ((unsigned int)f2bf(v4.w) << 16);
    *(uint2*)(Wb + i) = make_uint2(u01, u23);
}

// ---------------------------------------------------------------------------
// Kernel 3: fused  tanh(X_m @ W^T) -> softmax over m -> sum_m x*w -> *scaler
// Block: 256 threads (8 wave32), 64(b) x 64(e) output tile.
// Wave w: rows 16*(w>>1).., cols 32*(w&1).. -> 2 WMMA tiles x 6 modalities.
// LDS tiles padded to stride 40 shorts (80B rows: 16B-aligned frags, 20-bank
// row stride -> 16 distinct banks across the 16 row-lanes).
// ---------------------------------------------------------------------------
#define TSTR 40

__global__ __launch_bounds__(256) void fused_kernel(const float* __restrict__ x,
                                                    const unsigned short* __restrict__ Wb,
                                                    const float* __restrict__ scaler,
                                                    float* __restrict__ out) {
    __shared__ __align__(16) unsigned short lds_x[NMOD * 64 * TSTR]; // 30720 B
    __shared__ __align__(16) unsigned short lds_w[64 * TSTR];        //  5120 B  [e][k]

    const int tid  = threadIdx.x;
    const int e0   = blockIdx.x * 64;
    const int b0   = blockIdx.y * 64;
    const int wave = tid >> 5;
    const int lane = tid & 31;
    const int wr   = wave >> 1;   // 0..3 : b sub-block
    const int wc   = wave & 1;    // 0..1 : e sub-block
    const int half = lane >> 4;
    const int ln   = lane & 15;

    const v8f vzero = {0.f, 0.f, 0.f, 0.f, 0.f, 0.f, 0.f, 0.f};
    v8f acc[NMOD][2];
#pragma unroll
    for (int m = 0; m < NMOD; ++m) {
        acc[m][0] = vzero;
        acc[m][1] = vzero;
    }

    for (int k0 = 0; k0 < DIM; k0 += 32) {
        __syncthreads();
        // ---- stage W tile 64(e) x 32(k) bf16 via async global->LDS copy ----
        // (raw dword copy, no conversion: ideal for the CDNA5 async path)
#pragma unroll
        for (int i = 0; i < 4; ++i) {
            int d   = tid + i * 256;          // 1024 dwords total
            int row = d >> 4;                 // e within tile (16 dwords/row)
            int col = d & 15;                 // dword within row
            const unsigned int* g32 =
                (const unsigned int*)(Wb + (size_t)(e0 + row) * DIM + k0) + col;
            unsigned long long gaddr = (unsigned long long)(uintptr_t)g32;
            unsigned int ldsaddr =
                (unsigned int)(uintptr_t)&lds_w[row * TSTR + col * 2];
            asm volatile("global_load_async_to_lds_b32 %0, %1, off"
                         :
                         : "v"(ldsaddr), "v"(gaddr)
                         : "memory");
        }
        // ---- stage X tiles for all 6 modalities: 64(b) x 32(k), f32 -> bf16 ----
#pragma unroll
        for (int i = 0; i < 12; ++i) {
            int f   = tid + i * 256;          // 3072 float4 total
            int m   = f >> 9;
            int rem = f & 511;
            int row = rem >> 3;
            int kq  = rem & 7;
            const float4* src =
                (const float4*)(x + ((size_t)(m * BATCH + b0 + row)) * DIM + k0);
            float4 v4 = src[kq];
            if (k0 + 32 < DIM) __builtin_prefetch((const void*)(src + kq + 8), 0, 1);
            unsigned int u01 = (unsigned int)f2bf(v4.x) | ((unsigned int)f2bf(v4.y) << 16);
            unsigned int u23 = (unsigned int)f2bf(v4.z) | ((unsigned int)f2bf(v4.w) << 16);
            *(uint2*)&lds_x[(m * 64 + row) * TSTR + kq * 4] = make_uint2(u01, u23);
        }
        // async W-tile writes must land before the barrier releases readers
        asm volatile("s_wait_asynccnt 0x0" ::: "memory");
        __syncthreads();

        // ---- build B fragments (32x16 bf16) from [e][k] tile:
        //      lane col = ln, element j -> k = 16*half + j (contiguous pairs) ----
        Frag bfr[2];
#pragma unroll
        for (int t = 0; t < 2; ++t) {
            const unsigned short* ws = &lds_w[(32 * wc + 16 * t + ln) * TSTR];
#pragma unroll
            for (int v = 0; v < 8; ++v) {
                bfr[t].d[v] = *(const unsigned int*)&ws[16 * half + 2 * v];
            }
        }

        // ---- per modality: A fragment (16x32 bf16) + 2 WMMAs ----
#pragma unroll
        for (int m = 0; m < NMOD; ++m) {
            Frag a;
            const unsigned short* xs = &lds_x[(m * 64 + 16 * wr + ln) * TSTR];
#pragma unroll
            for (int v = 0; v < 8; ++v) {
                const int K0 = 16 * half + 2 * v;
                a.d[v] = *(const unsigned int*)&xs[K0];
            }
            acc[m][0] = __builtin_amdgcn_wmma_f32_16x16x32_bf16(
                false, a.v, false, bfr[0].v, (short)0, acc[m][0], false, false);
            acc[m][1] = __builtin_amdgcn_wmma_f32_16x16x32_bf16(
                false, a.v, false, bfr[1].v, (short)0, acc[m][1], false, false);
        }
    }

    // ---- epilogue: tanh -> softmax over m -> weighted sum of x -> *scaler ----
#pragma unroll
    for (int t = 0; t < 2; ++t) {
        const int e = e0 + 32 * wc + 16 * t + ln;
#pragma unroll
        for (int i = 0; i < 8; ++i) {
            const int b = b0 + 16 * wr + i + 8 * half;
            float s[NMOD];
            float mx = -3.402823e38f;
#pragma unroll
            for (int m = 0; m < NMOD; ++m) {
                float v = acc[m][t][i];
                v = fminf(fmaxf(v, -15.f), 15.f);
                float ex = __expf(2.f * v);
                s[m] = (ex - 1.f) / (ex + 1.f);   // tanh
                mx = fmaxf(mx, s[m]);
            }
            float w[NMOD];
            float den = 0.f;
#pragma unroll
            for (int m = 0; m < NMOD; ++m) {
                w[m] = __expf(s[m] - mx);
                den += w[m];
            }
            const float inv = 1.f / den;
            float o = 0.f;
#pragma unroll
            for (int m = 0; m < NMOD; ++m) {
                o += w[m] * x[((size_t)(m * BATCH + b)) * DIM + e];
            }
            out[(size_t)b * DIM + e] = o * inv * scaler[b];
        }
    }
}

// ---------------------------------------------------------------------------
extern "C" void kernel_launch(void* const* d_in, const int* in_sizes, int n_in,
                              void* d_out, int out_size, void* d_ws, size_t ws_size,
                              hipStream_t stream) {
    const float* x = (const float*)d_in[0];   // [6, 8192, 2048] f32
    const float* W = (const float*)d_in[1];   // [2048, 2048] f32
    float* out = (float*)d_out;               // [8192, 2048] f32

    // workspace layout: [0, 32KB) scaler[B] f32; [32KB, +8MB) Wb bf16 [D][D]
    float* scaler = (float*)d_ws;
    unsigned short* Wb = (unsigned short*)((char*)d_ws + 32768);

    scaler_kernel<<<BATCH, 256, 0, stream>>>(x, scaler);
    wconv_kernel<<<DIM * DIM / (256 * 4), 256, 0, stream>>>(W, Wb);
    fused_kernel<<<dim3(DIM / 64, BATCH / 64), 256, 0, stream>>>(x, Wb, scaler, out);
}
